// RawToSpectrograms_80865644249531
// MI455X (gfx1250) — matrix operand order repaired and checked by reading, not compiled, not run on hardware
//
#include <hip/hip_runtime.h>
#include <hip/hip_bf16.h>
#include <math.h>

// ---------------------------------------------------------------------------
// RawToSpectrograms for gfx1250 (MI455X)
//   - Only the frames actually sampled by the bilinear resize are gathered
//     (2 per output row), windowed, and stored to LDS as f16.
//   - rfft(256) expressed as real DFT GEMM executed with v_wmma_f32_16x16x32_f16.
//   - log|X| clipped, bilinear-resampled (half-pixel centers) to 128x64.
// ---------------------------------------------------------------------------

#define B_TOT      512        // 32 batches * 16 sensors
#define T_IN       160000
#define FRAME      256
#define HOP        128
#define NFRAMES    1250
#define NBINS      129
#define NBIN_PAD   144        // 9 tiles of 16
#define T_OUT      128
#define M_OUT      64
#define ROWS_PER_WG 16        // output rows per workgroup
#define NWAVES     9          // one wave per bin tile
#define NTHREADS   (NWAVES * 32)

typedef __attribute__((ext_vector_type(16))) _Float16 v16h;
typedef __attribute__((ext_vector_type(8)))  _Float16 v8h;
typedef __attribute__((ext_vector_type(8)))  float    v8f;

__constant__ float kScaleT = 1250.0f / 128.0f;   // 9.765625
__constant__ float kScaleM = 129.0f / 64.0f;     // 2.015625

// ---------------------------------------------------------------------------
// Init kernel: fill d_ws with the DFT basis, column-major ([bin][k], f16).
//   Bc[n][k] =  cos(2*pi*n*k/256)      (n <= 128, else 0)
//   Bs[n][k] = -sin(2*pi*n*k/256)      (n <= 128, else 0)
// Exact modular reduction of the phase keeps twiddles accurate.
// ---------------------------------------------------------------------------
__global__ void twiddle_init_kernel(_Float16* __restrict__ Bc,
                                    _Float16* __restrict__ Bs) {
    int idx = blockIdx.x * blockDim.x + threadIdx.x;
    const int per = NBIN_PAD * FRAME;           // 36864
    if (idx >= 2 * per) return;
    int which = idx / per;
    int r     = idx - which * per;
    int n = r >> 8;          // bin
    int k = r & 255;         // sample
    float v = 0.0f;
    if (n <= 128) {
        int ph = (n * k) & 255;                          // exact mod 256
        float ang = (float)ph * (float)(M_PI / 128.0);   // 2*pi*ph/256
        v = (which == 0) ? cosf(ang) : -sinf(ang);
    }
    ((which == 0) ? Bc : Bs)[r] = (_Float16)v;
}

// ---------------------------------------------------------------------------
// Main kernel: grid = 512 signals * 8 row-groups; 288 threads (9 waves).
// ---------------------------------------------------------------------------
__global__ __launch_bounds__(NTHREADS)
void spectrogram_kernel(const float* __restrict__ x,
                        const _Float16* __restrict__ Bc,
                        const _Float16* __restrict__ Bs,
                        float* __restrict__ out) {
    __shared__ __align__(32) _Float16 sF[2 * ROWS_PER_WG][FRAME];   // 16 KB
    __shared__ float sLM[2 * ROWS_PER_WG][NBIN_PAD + 1];            // ~18.6 KB

    const int tid   = threadIdx.x;
    const int bid   = blockIdx.x;
    const int sig   = bid >> 3;           // which (b,s) signal
    const int ibase = (bid & 7) * ROWS_PER_WG;
    const float* xs = x + (size_t)sig * T_IN;

    // ---- Stage 1: gather + window the 32 needed frames into LDS (f16) ----
    for (int idx = tid; idx < 2 * ROWS_PER_WG * FRAME; idx += NTHREADS) {
        int row = idx >> 8;               // 0..31  (frame slot)
        int k   = idx & 255;              // sample within frame
        int r   = row >> 1;               // output row within group
        int p   = row & 1;                // 0 -> t0, 1 -> t0+1
        float srcT = ((float)(ibase + r) + 0.5f) * kScaleT - 0.5f;
        int t0 = (int)floorf(srcT);
        int fr = t0 + p;
        if (fr > NFRAMES - 1) fr = NFRAMES - 1;
        if (fr < 0) fr = 0;
        int pos = fr * HOP + k;
        float v = (pos < T_IN) ? xs[pos] : 0.0f;
        // periodic Hann: 0.5 - 0.5*cos(2*pi*k/256)
        float w = 0.5f - 0.5f * __cosf((float)k * (float)(M_PI / 128.0));
        sF[row][k] = (_Float16)(v * w);
    }
    __syncthreads();

    // ---- Stage 2: real-DFT GEMM via WMMA --------------------------------
    const int wave = tid >> 5;            // bin tile 0..8
    const int lane = tid & 31;
    const int lo16 = lane & 15;
    const int half = lane >> 4;

    v8f accRe[2], accIm[2];
    accRe[0] = (v8f){0.f,0.f,0.f,0.f,0.f,0.f,0.f,0.f};
    accRe[1] = accRe[0];
    accIm[0] = accRe[0];
    accIm[1] = accRe[0];

    // B fragment: lane holds column n = wave*16+lo16; K chunk layout:
    //   element 2j,2j+1 <-> k = chunk*32 + 16*half + 2j  (contiguous 16 halves)
    const _Float16* bcRow = Bc + ((size_t)(wave * 16 + lo16)) * FRAME + 16 * half;
    const _Float16* bsRow = Bs + ((size_t)(wave * 16 + lo16)) * FRAME + 16 * half;
    // A fragment: lane holds frame m = mt*16+lo16; K layout contiguous:
    //   elements 0..15 <-> k = chunk*32 + 16*half .. +15
    const _Float16* aRow0 = &sF[lo16][16 * half];
    const _Float16* aRow1 = &sF[16 + lo16][16 * half];

    #pragma unroll
    for (int c = 0; c < FRAME / 32; ++c) {
        v16h bc = *(const v16h*)(bcRow + c * 32);
        v16h bs = *(const v16h*)(bsRow + c * 32);

        v8h a0lo = *(const v8h*)(aRow0 + c * 32);
        v8h a0hi = *(const v8h*)(aRow0 + c * 32 + 8);
        v8h a1lo = *(const v8h*)(aRow1 + c * 32);
        v8h a1hi = *(const v8h*)(aRow1 + c * 32 + 8);
        v16h a0 = __builtin_shufflevector(a0lo, a0hi,
                    0,1,2,3,4,5,6,7,8,9,10,11,12,13,14,15);
        v16h a1 = __builtin_shufflevector(a1lo, a1hi,
                    0,1,2,3,4,5,6,7,8,9,10,11,12,13,14,15);

        accRe[0] = __builtin_amdgcn_wmma_f32_16x16x32_f16(
                       false, a0, false, bc, (short)0, accRe[0], false, false);
        accIm[0] = __builtin_amdgcn_wmma_f32_16x16x32_f16(
                       false, a0, false, bs, (short)0, accIm[0], false, false);
        accRe[1] = __builtin_amdgcn_wmma_f32_16x16x32_f16(
                       false, a1, false, bc, (short)0, accRe[1], false, false);
        accIm[1] = __builtin_amdgcn_wmma_f32_16x16x32_f16(
                       false, a1, false, bs, (short)0, accIm[1], false, false);
    }

    // ---- Stage 3: clipped log-magnitude, scatter to LDS ------------------
    // C/D layout: row M = v + 8*half, col N = lane%16.
    const int bin = wave * 16 + lo16;
    #pragma unroll
    for (int mt = 0; mt < 2; ++mt) {
        const int frBase = mt * 16 + 8 * half;
        #pragma unroll
        for (int v = 0; v < 8; ++v) {
            float re = accRe[mt][v];
            float im = accIm[mt][v];
            float lm = logf(sqrtf(re * re + im * im) + 1e-9f);
            lm = fminf(fmaxf(lm, -12.0f), 12.0f);
            sLM[frBase + v][bin] = lm;
        }
    }
    __syncthreads();

    // ---- Stage 4: bilinear resample (half-pixel centers) + final clip ----
    for (int idx = tid; idx < ROWS_PER_WG * M_OUT; idx += NTHREADS) {
        int r = idx >> 6;
        int j = idx & 63;
        int i = ibase + r;

        float srcT = ((float)i + 0.5f) * kScaleT - 0.5f;
        float ft = srcT - floorf(srcT);

        float srcC = ((float)j + 0.5f) * kScaleM - 0.5f;
        int c0 = (int)floorf(srcC);
        float fc = srcC - (float)c0;
        if (c0 < 0) { c0 = 0; fc = 0.0f; }
        int c1 = c0 + 1;
        if (c1 > NBINS - 1) c1 = NBINS - 1;
        if (c0 > NBINS - 1) c0 = NBINS - 1;

        float a = sLM[2 * r    ][c0] * (1.0f - fc) + sLM[2 * r    ][c1] * fc;
        float b = sLM[2 * r + 1][c0] * (1.0f - fc) + sLM[2 * r + 1][c1] * fc;
        float val = a * (1.0f - ft) + b * ft;
        val = fminf(fmaxf(val, -12.0f), 12.0f);

        out[(((size_t)sig * T_OUT) + i) * M_OUT + j] = val;
    }
}

// ---------------------------------------------------------------------------
extern "C" void kernel_launch(void* const* d_in, const int* in_sizes, int n_in,
                              void* d_out, int out_size, void* d_ws, size_t ws_size,
                              hipStream_t stream) {
    const float* x = (const float*)d_in[0];
    float* out = (float*)d_out;

    _Float16* Bc = (_Float16*)d_ws;                     // 144*256 f16
    _Float16* Bs = Bc + NBIN_PAD * FRAME;               // 144*256 f16 (total 147 KB)

    const int twid_elems = 2 * NBIN_PAD * FRAME;
    twiddle_init_kernel<<<(twid_elems + 255) / 256, 256, 0, stream>>>(Bc, Bs);

    const int nblocks = B_TOT * (T_OUT / ROWS_PER_WG);  // 512 * 8 = 4096
    spectrogram_kernel<<<nblocks, NTHREADS, 0, stream>>>(x, Bc, Bs, out);
}
